// GPT_43662637531782
// MI455X (gfx1250) — compile-verified, matrix operand order checked by reading, not run on hardware
//
#include <hip/hip_runtime.h>

// ---------------------------------------------------------------------------
// GPT forward for MI455X (gfx1250, wave32, WMMA).
// GEMMs: bf16 A/B, f32 accumulate via v_wmma_f32_16x16x32_bf16.
//        B tiles staged in LDS via async copies (global_load_async_to_lds_b128,
//        ASYNCcnt); wave tile 32Mx64N -> 8 WMMA per 8 ds_load_b128.
// Attention: VALU + LDS softmax (1.4% of FLOPs, K/V L2-resident).
// lm_head (M=4): bandwidth-bound VALU dot kernel.
// ---------------------------------------------------------------------------

constexpr int Bc  = 4;
constexpr int Tc  = 1024;
constexpr int Dc  = 768;
constexpr int Hc  = 12;
constexpr int HDc = 64;
constexpr int Lc  = 4;
constexpr int Vc  = 50304;
constexpr int D3c = 3 * Dc;     // 2304
constexpr int D4c = 4 * Dc;     // 3072
constexpr int Mtok = Bc * Tc;   // 4096

typedef __attribute__((ext_vector_type(16))) __bf16 v16bf;
typedef __attribute__((ext_vector_type(8)))  __bf16 v8bf;
typedef __attribute__((ext_vector_type(8)))  float  v8f;

__device__ __forceinline__ __bf16 f2bf(float f) {
    unsigned u = __builtin_bit_cast(unsigned, f);
    unsigned r = (u + 0x7FFFu + ((u >> 16) & 1u)) >> 16;
    return __builtin_bit_cast(__bf16, (unsigned short)r);
}

// Load one 16-element bf16 fragment per lane following the ISA 16-bit A/B
// layout: lane holds K=[base..base+7] and K=[base+16..base+23] (two B128s).
// Works for both global and LDS (generic) pointers.
__device__ __forceinline__ v16bf load_frag16(const __bf16* p) {
    v8bf lo = *(const v8bf*)p;
    v8bf hi = *(const v8bf*)(p + 16);
    return __builtin_shufflevector(lo, hi, 0,1,2,3,4,5,6,7,8,9,10,11,12,13,14,15);
}

// CDNA5 async copy global -> LDS (tracked by ASYNCcnt). Inline asm because the
// clang builtin arity differs across toolchains. The LDS address operand is the
// low 32 bits of the generic pointer.
__device__ __forceinline__ void async_copy_b128(const __bf16* gsrc, unsigned lds_off) {
    asm volatile("global_load_async_to_lds_b128 %0, %1, off"
                 :: "v"(lds_off), "v"(gsrc) : "memory");
}
__device__ __forceinline__ void wait_async_le1() {
    asm volatile("s_wait_asynccnt 0x1" ::: "memory");
}
__device__ __forceinline__ void wait_async_le0() {
    asm volatile("s_wait_asynccnt 0x0" ::: "memory");
}

// ---------------------------------------------------------------------------
// WMMA GEMM: C[M,N] = A[M,K](bf16) * W[N,K]^T(bf16) + bias (+ resid)
// Block = 256 threads = 8 waves; block tile 256M x 64N; wave w owns rows
// [m0,m0+32) (two 16-row A fragments sharing the B fragments -> 8 WMMA/iter).
// B tile (64N x 32K) double-buffered in LDS via async copies, shared by all
// 8 waves. A fragments stream directly from global per wave.
// ---------------------------------------------------------------------------
template<int OUT_BF16, int RESID>
__global__ __launch_bounds__(256) void gemm_bf16_wmma(
    const __bf16* __restrict__ A,
    const __bf16* __restrict__ W,
    const float*  __restrict__ bias,
    const float*  __restrict__ resid,
    void* __restrict__ Cout, int M, int N, int K)
{
    __shared__ alignas(16) __bf16 Bsh[2][64][32];   // 2 x 4KB

    const int tid   = threadIdx.x;
    const int lane  = tid & 31;
    const int laneL = lane & 15;   // M (A) / N (B) index within tile
    const int laneH = lane >> 4;   // selects K halves per ISA layout
    const int wave  = tid >> 5;
    const int m0 = blockIdx.y * 256 + wave * 32;
    const int n0 = blockIdx.x * 64;

    // async-copy role: thread copies 16B of row (n0 + tid/4), chunk col (tid%4)*8
    const int crow  = tid >> 2;         // 0..63
    const int celt  = (tid & 3) * 8;    // element offset within 32-wide chunk
    const __bf16* gW = W + (size_t)(n0 + crow) * K + celt;

    const __bf16* Ap0 = A + (size_t)(m0 +  0 + laneL) * K + laneH * 8;
    const __bf16* Ap1 = A + (size_t)(m0 + 16 + laneL) * K + laneH * 8;

    v8f acc[2][4];
    #pragma unroll
    for (int f = 0; f < 2; ++f)
        #pragma unroll
        for (int s = 0; s < 4; ++s)
            acc[f][s] = v8f{0,0,0,0,0,0,0,0};

    // prologue: fill buffer 0
    async_copy_b128(gW, (unsigned)(uintptr_t)&Bsh[0][crow][celt]);

    int buf = 0;
    for (int kk = 0; kk < K - 32; kk += 32) {
        // steady state: branch-free issue-next / wait-current / compute
        async_copy_b128(gW + kk + 32, (unsigned)(uintptr_t)&Bsh[buf ^ 1][crow][celt]);
        wait_async_le1();               // our copy for current buffer is done
        __syncthreads();                // everyone's copy landed in LDS

        __builtin_prefetch(Ap0 + kk + 128, 0, 1);   // global_prefetch_b8
        __builtin_prefetch(Ap1 + kk + 128, 0, 1);
        v16bf a0 = load_frag16(Ap0 + kk);
        v16bf a1 = load_frag16(Ap1 + kk);
        v16bf b0 = load_frag16(&Bsh[buf][ 0 + laneL][laneH * 8]);
        v16bf b1 = load_frag16(&Bsh[buf][16 + laneL][laneH * 8]);
        v16bf b2 = load_frag16(&Bsh[buf][32 + laneL][laneH * 8]);
        v16bf b3 = load_frag16(&Bsh[buf][48 + laneL][laneH * 8]);
        acc[0][0] = __builtin_amdgcn_wmma_f32_16x16x32_bf16(false, a0, false, b0, (short)0, acc[0][0], false, false);
        acc[0][1] = __builtin_amdgcn_wmma_f32_16x16x32_bf16(false, a0, false, b1, (short)0, acc[0][1], false, false);
        acc[0][2] = __builtin_amdgcn_wmma_f32_16x16x32_bf16(false, a0, false, b2, (short)0, acc[0][2], false, false);
        acc[0][3] = __builtin_amdgcn_wmma_f32_16x16x32_bf16(false, a0, false, b3, (short)0, acc[0][3], false, false);
        acc[1][0] = __builtin_amdgcn_wmma_f32_16x16x32_bf16(false, a1, false, b0, (short)0, acc[1][0], false, false);
        acc[1][1] = __builtin_amdgcn_wmma_f32_16x16x32_bf16(false, a1, false, b1, (short)0, acc[1][1], false, false);
        acc[1][2] = __builtin_amdgcn_wmma_f32_16x16x32_bf16(false, a1, false, b2, (short)0, acc[1][2], false, false);
        acc[1][3] = __builtin_amdgcn_wmma_f32_16x16x32_bf16(false, a1, false, b3, (short)0, acc[1][3], false, false);

        __syncthreads();                // protect buffer before next overwrite
        buf ^= 1;
    }

    // peeled final chunk
    {
        const int kk = K - 32;
        wait_async_le0();
        __syncthreads();
        v16bf a0 = load_frag16(Ap0 + kk);
        v16bf a1 = load_frag16(Ap1 + kk);
        v16bf b0 = load_frag16(&Bsh[buf][ 0 + laneL][laneH * 8]);
        v16bf b1 = load_frag16(&Bsh[buf][16 + laneL][laneH * 8]);
        v16bf b2 = load_frag16(&Bsh[buf][32 + laneL][laneH * 8]);
        v16bf b3 = load_frag16(&Bsh[buf][48 + laneL][laneH * 8]);
        acc[0][0] = __builtin_amdgcn_wmma_f32_16x16x32_bf16(false, a0, false, b0, (short)0, acc[0][0], false, false);
        acc[0][1] = __builtin_amdgcn_wmma_f32_16x16x32_bf16(false, a0, false, b1, (short)0, acc[0][1], false, false);
        acc[0][2] = __builtin_amdgcn_wmma_f32_16x16x32_bf16(false, a0, false, b2, (short)0, acc[0][2], false, false);
        acc[0][3] = __builtin_amdgcn_wmma_f32_16x16x32_bf16(false, a0, false, b3, (short)0, acc[0][3], false, false);
        acc[1][0] = __builtin_amdgcn_wmma_f32_16x16x32_bf16(false, a1, false, b0, (short)0, acc[1][0], false, false);
        acc[1][1] = __builtin_amdgcn_wmma_f32_16x16x32_bf16(false, a1, false, b1, (short)0, acc[1][1], false, false);
        acc[1][2] = __builtin_amdgcn_wmma_f32_16x16x32_bf16(false, a1, false, b2, (short)0, acc[1][2], false, false);
        acc[1][3] = __builtin_amdgcn_wmma_f32_16x16x32_bf16(false, a1, false, b3, (short)0, acc[1][3], false, false);
    }

    #pragma unroll
    for (int f = 0; f < 2; ++f) {
        #pragma unroll
        for (int s = 0; s < 4; ++s) {
            int n = n0 + s * 16 + laneL;
            float bv = bias ? bias[n] : 0.0f;
            #pragma unroll
            for (int i = 0; i < 8; ++i) {
                // C/D layout: VGPR i holds row (tilebase + i + 8*(lane/16))
                int m = m0 + f * 16 + laneH * 8 + i;
                float v = acc[f][s][i] + bv;
                if (RESID) v += resid[(size_t)m * N + n];
                if (OUT_BF16) ((__bf16*)Cout)[(size_t)m * N + n] = f2bf(v);
                else          ((float* )Cout)[(size_t)m * N + n] = v;
            }
        }
    }
}

// ---------------------------------------------------------------------------
// LayerNorm: one block per row of D=768; OUT_BF16 for GEMM inputs, f32 for lnf.
// Row r reads x[(r*row_step + row_off)*D], writes out[r*D].
// ---------------------------------------------------------------------------
template<int OUT_BF16>
__global__ __launch_bounds__(256) void layernorm_kernel(
    const float* __restrict__ x, const float* __restrict__ w,
    const float* __restrict__ b, void* __restrict__ out,
    int row_step, int row_off)
{
    __shared__ float red[256];
    const int r = blockIdx.x;
    const int tid = threadIdx.x;
    const float* xr = x + (size_t)((size_t)r * row_step + row_off) * Dc;

    float s = 0.f;
    for (int d = tid; d < Dc; d += 256) s += xr[d];
    red[tid] = s; __syncthreads();
    for (int st = 128; st > 0; st >>= 1) { if (tid < st) red[tid] += red[tid + st]; __syncthreads(); }
    const float mu = red[0] / Dc;
    __syncthreads();

    float s2 = 0.f;
    for (int d = tid; d < Dc; d += 256) { float t = xr[d] - mu; s2 += t * t; }
    red[tid] = s2; __syncthreads();
    for (int st = 128; st > 0; st >>= 1) { if (tid < st) red[tid] += red[tid + st]; __syncthreads(); }
    const float rstd = rsqrtf(red[0] / Dc + 1e-5f);

    for (int d = tid; d < Dc; d += 256) {
        float v = (xr[d] - mu) * rstd * w[d] + b[d];
        if (OUT_BF16) ((__bf16*)out)[(size_t)r * Dc + d] = f2bf(v);
        else          ((float* )out)[(size_t)r * Dc + d] = v;
    }
}

// ---------------------------------------------------------------------------
// Causal attention, one block per (b, h, query row). Two-pass softmax in LDS.
// qkv: [B,T,3D] f32 (q|k|v). Writes y as bf16 [B,T,D] for the proj GEMM.
// ---------------------------------------------------------------------------
__global__ __launch_bounds__(256) void attention_kernel(
    const float* __restrict__ qkv, __bf16* __restrict__ ybf)
{
    __shared__ float qs[HDc];
    __shared__ float sc[Tc];
    __shared__ float red[256];
    __shared__ float osh[256];

    const int bh  = blockIdx.y;
    const int bq  = blockIdx.x;          // query row
    const int bb  = bh / Hc;
    const int hh  = bh % Hc;
    const int tid = threadIdx.x;

    if (tid < HDc)
        qs[tid] = qkv[(size_t)(bb * Tc + bq) * D3c + hh * HDc + tid] * 0.125f; // 1/sqrt(64)
    __syncthreads();

    const int nk = bq + 1;               // causal: keys 0..bq
    for (int j = tid; j < nk; j += 256) {
        const float* kr = qkv + (size_t)(bb * Tc + j) * D3c + Dc + hh * HDc;
        float s = 0.f;
        #pragma unroll
        for (int d = 0; d < HDc; ++d) s += qs[d] * kr[d];
        sc[j] = s;
    }
    __syncthreads();

    float lm = -3.0e38f;
    for (int j = tid; j < nk; j += 256) lm = fmaxf(lm, sc[j]);
    red[tid] = lm; __syncthreads();
    for (int st = 128; st > 0; st >>= 1) { if (tid < st) red[tid] = fmaxf(red[tid], red[tid + st]); __syncthreads(); }
    const float mx = red[0];
    __syncthreads();

    float ls = 0.f;
    for (int j = tid; j < nk; j += 256) { float p = __expf(sc[j] - mx); sc[j] = p; ls += p; }
    red[tid] = ls; __syncthreads();
    for (int st = 128; st > 0; st >>= 1) { if (tid < st) red[tid] += red[tid + st]; __syncthreads(); }
    const float inv = 1.f / red[0];

    // O = P * V : 4 chunks over keys, 64 dims
    const int c = tid >> 6, d = tid & 63;
    float o = 0.f;
    for (int j = c; j < nk; j += 4)
        o += sc[j] * qkv[(size_t)(bb * Tc + j) * D3c + 2 * Dc + hh * HDc + d];
    osh[tid] = o; __syncthreads();

    if (tid < HDc) {
        float r = (osh[tid] + osh[64 + tid] + osh[128 + tid] + osh[192 + tid]) * inv;
        ybf[(size_t)(bb * Tc + bq) * Dc + hh * HDc + tid] = f2bf(r);
    }
}

// ---------------------------------------------------------------------------
// x = wte[idx] + wpe
// ---------------------------------------------------------------------------
__global__ __launch_bounds__(256) void embed_kernel(
    const int* __restrict__ idx, const float* __restrict__ wte,
    const float* __restrict__ wpe, float* __restrict__ x)
{
    size_t i = (size_t)blockIdx.x * 256 + threadIdx.x;
    int d = (int)(i % Dc);
    size_t bt = i / Dc;
    int t = (int)(bt % Tc);
    x[i] = wte[(size_t)idx[bt] * Dc + d] + wpe[(size_t)t * Dc + d];
}

// ---------------------------------------------------------------------------
// sw = silu(s1) * s2   -> bf16 for the mlp_proj GEMM
// ---------------------------------------------------------------------------
__global__ __launch_bounds__(256) void swiglu_kernel(
    const float* __restrict__ s1, const float* __restrict__ s2,
    __bf16* __restrict__ out, size_t n)
{
    size_t i = (size_t)blockIdx.x * 256 + threadIdx.x;
    if (i >= n) return;
    float a = s1[i];
    float sg = a / (1.f + __expf(-a));
    out[i] = f2bf(sg * s2[i]);
}

// f32 -> bf16, weights stored [N,K] already
__global__ __launch_bounds__(256) void cvt_bf16_kernel(
    const float* __restrict__ in, __bf16* __restrict__ out, size_t n)
{
    size_t i = (size_t)blockIdx.x * 256 + threadIdx.x;
    if (i < n) out[i] = f2bf(in[i]);
}

// Tiled transpose-convert: in [K,N] f32 -> out [N,K] bf16 (for sw_W / sw_V)
__global__ __launch_bounds__(256) void transpose_cvt_kernel(
    const float* __restrict__ in, __bf16* __restrict__ out, int K, int N)
{
    __shared__ float tile[32][33];
    const int k0 = blockIdx.y * 32, n0 = blockIdx.x * 32;
    for (int yy = threadIdx.y; yy < 32; yy += 8)
        tile[yy][threadIdx.x] = in[(size_t)(k0 + yy) * N + n0 + threadIdx.x];
    __syncthreads();
    for (int yy = threadIdx.y; yy < 32; yy += 8)
        out[(size_t)(n0 + yy) * K + k0 + threadIdx.x] = f2bf(tile[threadIdx.x][yy]);
}

// ---------------------------------------------------------------------------
// logits[b, v] = dot(xf[b,:], lm_head_w[v,:]); one wave per (b,v), M=4 so
// this is pure HBM bandwidth -> no WMMA, no weight conversion.
// ---------------------------------------------------------------------------
__global__ __launch_bounds__(256) void lm_head_kernel(
    const float* __restrict__ xf, const float* __restrict__ Wv,
    float* __restrict__ out)
{
    const int lane = threadIdx.x & 31;
    const int wid  = (int)(blockIdx.x * 8 + (threadIdx.x >> 5));
    const int v = wid % Vc;
    const int b = wid / Vc;
    float s = 0.f;
    const float* xr = xf + (size_t)b * Dc;
    const float* wr = Wv + (size_t)v * Dc;
    for (int d = lane; d < Dc; d += 32) s += xr[d] * wr[d];
    #pragma unroll
    for (int m = 16; m > 0; m >>= 1) s += __shfl_xor(s, m, 32);
    if (lane == 0) out[(size_t)b * Vc + v] = s;
}

// ---------------------------------------------------------------------------
// Host orchestration
// ---------------------------------------------------------------------------
extern "C" void kernel_launch(void* const* d_in, const int* in_sizes, int n_in,
                              void* d_out, int out_size, void* d_ws, size_t ws_size,
                              hipStream_t stream)
{
    (void)in_sizes; (void)n_in; (void)out_size; (void)ws_size;

    const int*   idx        = (const int*)  d_in[0];
    const float* wte        = (const float*)d_in[1];
    const float* wpe        = (const float*)d_in[2];
    const float* ln1_w      = (const float*)d_in[3];
    const float* ln1_b      = (const float*)d_in[4];
    const float* c_attn_w   = (const float*)d_in[5];
    const float* c_attn_b   = (const float*)d_in[6];
    const float* c_proj_w   = (const float*)d_in[7];
    const float* c_proj_b   = (const float*)d_in[8];
    const float* ln2_w      = (const float*)d_in[9];
    const float* ln2_b      = (const float*)d_in[10];
    const float* c_fc_w     = (const float*)d_in[11];
    const float* c_fc_b     = (const float*)d_in[12];
    const float* sw_W       = (const float*)d_in[13];
    const float* sw_V       = (const float*)d_in[14];
    const float* sw_b       = (const float*)d_in[15];
    const float* sw_c       = (const float*)d_in[16];
    const float* mlp_proj_w = (const float*)d_in[17];
    const float* mlp_proj_b = (const float*)d_in[18];
    const float* lnf_w      = (const float*)d_in[19];
    const float* lnf_b      = (const float*)d_in[20];
    const float* lm_head_w  = (const float*)d_in[21];

    char* ws = (char*)d_ws;
    size_t off = 0;
    auto carve = [&](size_t bytes) -> char* {
        char* p = ws + off;
        off += (bytes + 255) & ~(size_t)255;
        return p;
    };

    float*  x_f   = (float*) carve((size_t)Mtok * Dc  * 4);
    __bf16* xn    = (__bf16*)carve((size_t)Mtok * Dc  * 2);
    float*  qkv_f = (float*) carve((size_t)Mtok * D3c * 4);
    __bf16* y_bf  = (__bf16*)carve((size_t)Mtok * Dc  * 2);
    __bf16* h_bf  = (__bf16*)carve((size_t)Mtok * D4c * 2);
    float*  s1_f  = (float*) carve((size_t)Mtok * D4c * 4);
    float*  s2_f  = (float*) carve((size_t)Mtok * D4c * 4);
    __bf16* swbf  = (__bf16*)carve((size_t)Mtok * D4c * 2);
    __bf16* wq    = (__bf16*)carve((size_t)D3c * Dc  * 2);
    __bf16* wp    = (__bf16*)carve((size_t)Dc  * Dc  * 2);
    __bf16* wf    = (__bf16*)carve((size_t)D4c * Dc  * 2);
    __bf16* w1    = (__bf16*)carve((size_t)D4c * D4c * 2);
    __bf16* w2    = (__bf16*)carve((size_t)D4c * D4c * 2);
    __bf16* wm    = (__bf16*)carve((size_t)Dc  * D4c * 2);
    float*  xf    = (float*) carve((size_t)Bc * Dc * 4);

    // token + position embedding
    embed_kernel<<<((size_t)Mtok * Dc) / 256, 256, 0, stream>>>(idx, wte, wpe, x_f);

    const dim3 gN3 (D3c / 64, Mtok / 256);   // N = 2304
    const dim3 gND (Dc  / 64, Mtok / 256);   // N = 768
    const dim3 gN4 (D4c / 64, Mtok / 256);   // N = 3072

    for (int l = 0; l < Lc; ++l) {
        const float* caw = c_attn_w   + (size_t)l * D3c * Dc;
        const float* cpw = c_proj_w   + (size_t)l * Dc  * Dc;
        const float* cfw = c_fc_w     + (size_t)l * D4c * Dc;
        const float* sWl = sw_W       + (size_t)l * D4c * D4c;
        const float* sVl = sw_V       + (size_t)l * D4c * D4c;
        const float* mpw = mlp_proj_w + (size_t)l * Dc  * D4c;

        // weight conversion into per-layer bf16 staging
        cvt_bf16_kernel<<<((size_t)D3c * Dc + 255) / 256, 256, 0, stream>>>(caw, wq, (size_t)D3c * Dc);
        cvt_bf16_kernel<<<((size_t)Dc  * Dc + 255) / 256, 256, 0, stream>>>(cpw, wp, (size_t)Dc  * Dc);
        cvt_bf16_kernel<<<((size_t)D4c * Dc + 255) / 256, 256, 0, stream>>>(cfw, wf, (size_t)D4c * Dc);
        cvt_bf16_kernel<<<((size_t)Dc * D4c + 255) / 256, 256, 0, stream>>>(mpw, wm, (size_t)Dc * D4c);
        // sw_W / sw_V are [K,N] in the einsum -> transpose into [N,K]
        transpose_cvt_kernel<<<dim3(D4c / 32, D4c / 32), dim3(32, 8), 0, stream>>>(sWl, w1, D4c, D4c);
        transpose_cvt_kernel<<<dim3(D4c / 32, D4c / 32), dim3(32, 8), 0, stream>>>(sVl, w2, D4c, D4c);

        // attention
        layernorm_kernel<1><<<Mtok, 256, 0, stream>>>(x_f, ln1_w + (size_t)l * Dc, ln1_b + (size_t)l * Dc, xn, 1, 0);
        gemm_bf16_wmma<0, 0><<<gN3, 256, 0, stream>>>(xn, wq, c_attn_b + (size_t)l * D3c, nullptr,
                                                      qkv_f, Mtok, D3c, Dc);
        attention_kernel<<<dim3(Tc, Bc * Hc), 256, 0, stream>>>(qkv_f, y_bf);
        gemm_bf16_wmma<0, 1><<<gND, 256, 0, stream>>>(y_bf, wp, c_proj_b + (size_t)l * Dc, x_f,
                                                      x_f, Mtok, Dc, Dc);   // x += proj(y)

        // SwiGLU MLP
        layernorm_kernel<1><<<Mtok, 256, 0, stream>>>(x_f, ln2_w + (size_t)l * Dc, ln2_b + (size_t)l * Dc, xn, 1, 0);
        gemm_bf16_wmma<1, 0><<<gN4, 256, 0, stream>>>(xn, wf, c_fc_b + (size_t)l * D4c, nullptr,
                                                      h_bf, Mtok, D4c, Dc);
        gemm_bf16_wmma<0, 0><<<gN4, 256, 0, stream>>>(h_bf, w1, sw_b + (size_t)l * D4c, nullptr,
                                                      s1_f, Mtok, D4c, D4c);
        gemm_bf16_wmma<0, 0><<<gN4, 256, 0, stream>>>(h_bf, w2, sw_c + (size_t)l * D4c, nullptr,
                                                      s2_f, Mtok, D4c, D4c);
        swiglu_kernel<<<((size_t)Mtok * D4c + 255) / 256, 256, 0, stream>>>(s1_f, s2_f, swbf, (size_t)Mtok * D4c);
        gemm_bf16_wmma<0, 1><<<gND, 256, 0, stream>>>(swbf, wm, mlp_proj_b + (size_t)l * Dc, x_f,
                                                      x_f, Mtok, Dc, D4c); // x += mlp
    }

    // final LN on last position of each batch row, then logits
    layernorm_kernel<0><<<Bc, 256, 0, stream>>>(x_f, lnf_w, lnf_b, xf, Tc, Tc - 1);
    lm_head_kernel<<<(Bc * Vc) / 8, 256, 0, stream>>>(xf, lm_head_w, (float*)d_out);
}
